// LstmCell_24309514895774
// MI455X (gfx1250) — matrix operand order repaired and checked by reading, not compiled, not run on hardware
//
#include <hip/hip_runtime.h>
#include <hip/hip_bf16.h>

// ---------------------------------------------------------------------------
// Fused LSTM-variant cell for MI455X (gfx1250, wave32, WMMA).
//   gates = [x | h] @ W + b ; i=sig(g0), j=tanh(g1), o=sig(g2)
//   new_c = (1-i)*c + i*j ; new_h = tanh(new_c)*o
// Single fused kernel: bf16 WMMA GEMM (fp32 accum) + elementwise epilogue.
// ---------------------------------------------------------------------------

typedef __attribute__((ext_vector_type(16))) __bf16 v16bf;
typedef __attribute__((ext_vector_type(8)))  float  v8f;

#define B_ROWS 16384
#define IN_DIM 1024
#define H_DIM  1024
#define K_DIM  (IN_DIM + H_DIM)   // 2048
#define NG     (3 * H_DIM)        // 3072

#define BM 128            // block tile rows
#define BN 64             // block tile cols (per gate)
#define KT 32             // K per WMMA step (bf16 16x16x32)
#define NKT (K_DIM / KT)  // 64 steps
// Stride 20 uints per row/col: keeps 16B alignment for ds_load_b128 AND
// tiles all 64 LDS banks (20*r mod 64 hits every multiple of 4, each lane's
// b128 then covers its own 4-bank slice).
#define A_STRIDE 20
#define B_STRIDE 20

__device__ __forceinline__ unsigned pack2bf(float a, float b) {
  __hip_bfloat162 t = __float22bfloat162_rn(make_float2(a, b));  // v_cvt_pk_bf16_f32
  unsigned r;
  __builtin_memcpy(&r, &t, sizeof(r));
  return r;
}
__device__ __forceinline__ float sigmoidf(float v) {
  return 1.0f / (1.0f + __expf(-v));
}

union AFrag {
  v16bf    v;
  uint4    q[2];
  unsigned u[8];
};

__global__ __launch_bounds__(256)
void lstm_cell_fused_wmma(const float* __restrict__ x,
                          const float* __restrict__ hin,
                          const float* __restrict__ cin,
                          const float* __restrict__ W,
                          const float* __restrict__ bias,
                          float* __restrict__ out) {
  // Double-buffered bf16 tiles (packed pairs in uints). 51200 B total.
  __shared__ unsigned Abuf[2][BM * A_STRIDE];
  __shared__ unsigned Bbuf[2][3][BN * B_STRIDE];

  const int tid   = threadIdx.x;
  const int lane  = tid & 31;
  const int wave  = tid >> 5;       // 0..7
  const int wm    = wave & 3;       // 4 waves along M
  const int wn    = wave >> 2;      // 2 waves along N
  const int lmod  = lane & 15;
  const int lhalf = lane >> 4;

  const int blockM = blockIdx.x * BM;
  const int blockN = blockIdx.y * BN;

  v8f acc[3][2][2];
#pragma unroll
  for (int g = 0; g < 3; ++g)
#pragma unroll
    for (int i = 0; i < 2; ++i)
#pragma unroll
      for (int j = 0; j < 2; ++j)
        acc[g][i][j] = (v8f){0.f, 0.f, 0.f, 0.f, 0.f, 0.f, 0.f, 0.f};

  // --- LDS fill: A tile (BM x KT) fp32 -> bf16 pairs, row-major.
  // 1024 % KT == 0 so each K-slab is wholly from x or wholly from h.
  auto fillA = [&](int buf, int t) {
    const int kk = t * KT;
    const float* src = (kk < IN_DIM) ? (x + kk) : (hin + (kk - IN_DIM));
    const int row = tid >> 1;              // 0..127
    const int kh  = (tid & 1) << 4;        // 0 or 16
    const float* p = src + (size_t)(blockM + row) * (size_t)IN_DIM + kh;
    unsigned* dst = &Abuf[buf][row * A_STRIDE + (kh >> 1)];
#pragma unroll
    for (int j = 0; j < 8; ++j)
      dst[j] = pack2bf(p[2 * j], p[2 * j + 1]);   // 8 contiguous uints -> b128 stores
  };

  // --- LDS fill: three B tiles (KT x BN), column-major K-pairs.
  // Each thread owns K-pairs of one column: 2 coalesced row reads ->
  // 1 packed cvt -> 1 b32 DS store.
  auto fillB = [&](int buf, int t) {
    const int kk = t * KT;
    const int cc = tid & 63;   // column within tile
    const int kq = tid >> 6;   // 0..3
#pragma unroll
    for (int g = 0; g < 3; ++g) {
      unsigned* bp = &Bbuf[buf][g][cc * B_STRIDE];
      const float* wp = W + (size_t)kk * (size_t)NG
                          + (size_t)(g * H_DIM + blockN + cc);
#pragma unroll
      for (int p = 0; p < 4; ++p) {
        const int kpair = p * 4 + kq;              // 0..15
        const float f0 = wp[(size_t)(2 * kpair) * (size_t)NG];
        const float f1 = wp[(size_t)(2 * kpair + 1) * (size_t)NG];
        bp[kpair] = pack2bf(f0, f1);
      }
    }
  };

  // --- One K-step: preload ALL fragments (b128 DS loads), then 12 WMMAs.
  auto computeStep = [&](int buf) {
    AFrag a[2];
    AFrag bb[3][2];
    // A frag (ISA 7.12.2, 16-bit A 16x32): lane L<16 -> M=L, VGPR r holds
    // K = 2*(r%4) + 16*(r/4) (+8 for L>=16). u[0..3] and u[4..7] contiguous.
#pragma unroll
    for (int ms = 0; ms < 2; ++ms) {
      const uint4* ap = reinterpret_cast<const uint4*>(
          &Abuf[buf][(wm * 32 + ms * 16 + lmod) * A_STRIDE + lhalf * 4]);
      a[ms].q[0] = ap[0];
      a[ms].q[1] = ap[2];   // +8 uints
    }
    // B frag (16-bit B 32x16): same K order as A: VGPR r holds K = 2*(r%4) + 16*(r/4) (+8 for lanes>=16).
#pragma unroll
    for (int g = 0; g < 3; ++g)
#pragma unroll
      for (int ns = 0; ns < 2; ++ns) {
        const uint4* bp = reinterpret_cast<const uint4*>(
            &Bbuf[buf][g][(wn * 32 + ns * 16 + lmod) * B_STRIDE + lhalf * 4]);
        bb[g][ns].q[0] = bp[0];
        bb[g][ns].q[1] = bp[2];
      }
#pragma unroll
    for (int g = 0; g < 3; ++g)
#pragma unroll
      for (int ns = 0; ns < 2; ++ns)
#pragma unroll
        for (int ms = 0; ms < 2; ++ms)
          acc[g][ms][ns] = __builtin_amdgcn_wmma_f32_16x16x32_bf16(
              false, a[ms].v, false, bb[g][ns].v, (short)0, acc[g][ms][ns],
              false, false);
  };

  // --- Main double-buffered K loop.
  fillA(0, 0);
  fillB(0, 0);
  __syncthreads();
#pragma unroll 1
  for (int t = 0; t < NKT; ++t) {
    const int cur = t & 1;
    if (t + 1 < NKT) {
      fillA(cur ^ 1, t + 1);
      fillB(cur ^ 1, t + 1);
    }
    computeStep(cur);
    __syncthreads();
  }

  // --- Fused epilogue. C/D layout: VGPR r, lanes 0-15 -> row r, col lane;
  // lanes 16-31 -> row 8+r, col lane-16.
  float* out_h = out;
  float* out_c = out + (size_t)B_ROWS * (size_t)H_DIM;
#pragma unroll
  for (int ms = 0; ms < 2; ++ms) {
#pragma unroll
    for (int ns = 0; ns < 2; ++ns) {
      const int n = blockN + wn * 32 + ns * 16 + lmod;   // column in H
      const float bi = bias[n];
      const float bj = bias[H_DIM + n];
      const float bo = bias[2 * H_DIM + n];
      const int mbase = blockM + wm * 32 + ms * 16 + lhalf * 8;
      const v8f gi = acc[0][ms][ns];
      const v8f gj = acc[1][ms][ns];
      const v8f go = acc[2][ms][ns];
#pragma unroll
      for (int r = 0; r < 8; ++r) {
        const size_t idx = (size_t)(mbase + r) * (size_t)H_DIM + (size_t)n;
        const float fi = sigmoidf(gi[r] + bi);
        const float fj = tanhf(gj[r] + bj);
        const float fo = sigmoidf(go[r] + bo);
        const float cv = cin[idx];
        const float nc = (1.0f - fi) * cv + fi * fj;
        out_c[idx] = nc;
        out_h[idx] = tanhf(nc) * fo;
      }
    }
  }
}

extern "C" void kernel_launch(void* const* d_in, const int* in_sizes, int n_in,
                              void* d_out, int out_size, void* d_ws,
                              size_t ws_size, hipStream_t stream) {
  (void)in_sizes; (void)n_in; (void)out_size; (void)d_ws; (void)ws_size;
  const float* x    = (const float*)d_in[0];
  const float* hin  = (const float*)d_in[1];
  const float* cin  = (const float*)d_in[2];
  const float* W    = (const float*)d_in[3];
  const float* bias = (const float*)d_in[4];
  dim3 grid(B_ROWS / BM, H_DIM / BN);   // 128 x 16 = 2048 blocks
  lstm_cell_fused_wmma<<<grid, 256, 0, stream>>>(x, hin, cin, W, bias,
                                                 (float*)d_out);
}
